// GNN_55430847922289
// MI455X (gfx1250) — matrix operand order repaired and checked by reading, not compiled, not run on hardware
//
#include <hip/hip_runtime.h>
#include <hip/hip_bf16.h>
#include <math.h>

// ---------------- static problem sizes (match reference) ----------------
#define BGR   64
#define NN0   128
#define EPG   1024
#define ETOT  (BGR*EPG)
#define NTOT  (BGR*NN0)
#define F0    30
#define ED    11
#define HH    4
#define CC    128
#define HC    512
#define EMB   128

typedef __bf16 v16bf __attribute__((ext_vector_type(16)));
typedef float  v8f   __attribute__((ext_vector_type(8)));
typedef unsigned int u32x4 __attribute__((ext_vector_type(4)));

// fp32 -> bf16 RNE without float->__bf16 casts
__device__ __forceinline__ unsigned short f2us(float f) {
    union { float f; unsigned u; } x; x.f = f;
    return (unsigned short)((x.u + 0x7FFFu + ((x.u >> 16) & 1u)) >> 16);
}

// IEEE-ordered atomic float max (int/uint trick)
__device__ __forceinline__ void atomicMaxF(float* a, float v) {
    if (v >= 0.0f) atomicMax((int*)a, __float_as_int(v));
    else           atomicMin((unsigned int*)a, __float_as_uint(v));
}

__device__ __forceinline__ float edge_proj(const float* __restrict__ er,
                                           const float* __restrict__ we, int col) {
    float s = 0.f;
#pragma unroll
    for (int d = 0; d < ED; ++d) s += er[d] * we[d * HC + col];
    return s;
}

// ---------------- conversion passes (run once, outside hot loops) -------
__global__ void gnn_cvt_bf16(const float* __restrict__ s, unsigned short* __restrict__ d, int n) {
    int i = blockIdx.x * blockDim.x + threadIdx.x;
    if (i < n) d[i] = f2us(s[i]);
}
// pad K -> Kp (multiple of 32) with zeros so GEMM A loads are unguarded b128
__global__ void gnn_cvt_pad_bf16(const float* __restrict__ s, unsigned short* __restrict__ d,
                                 int M, int K, int Kp) {
    int i = blockIdx.x * blockDim.x + threadIdx.x;
    if (i >= M * Kp) return;
    int r = i / Kp, c = i - r * Kp;
    d[i] = (c < K) ? f2us(s[(size_t)r * K + c]) : (unsigned short)0;
}

// ---------------- bf16 WMMA GEMM: C[M,N] = act(A@W + bias) ----------------
// A: bf16 [M, lda] (lda = K padded to mult of 32). W: bf16 [K, N] row-major.
// block (32,4): 4 waves; each wave one 16-row tile, all share a 64-col strip.
// Per K-step: stage 32x64 B tile transposed in LDS; 1 A-frag (2x b128 global),
// all 4 B-frags preloaded (8x ds_load_b128, one wait), then 4 back-to-back
// WMMAs reusing the A fragment. Requires M%64==0, N%64==0.
__global__ void gnn_wmma_gemm_bf16(const unsigned short* __restrict__ A, int lda,
                                   const unsigned short* __restrict__ W,
                                   const float* __restrict__ bias,
                                   float* __restrict__ C,
                                   int M, int K, int N, int act) {
    __shared__ unsigned short sBT[64 * 40];   // [n][k] transposed, 80B row stride
    const int tileN = blockIdx.x;             // 64-col strip
    const int tileM = blockIdx.y * 4 + threadIdx.y;
    const int lane  = threadIdx.x;
    const int half  = lane >> 4;
    const int l15   = lane & 15;
    const int arow  = tileM * 16 + l15;
    const int tid   = threadIdx.y * 32 + lane;
    const int sk    = tid >> 2;               // K row staged by this thread (0..31)
    const int sn0   = (tid & 3) * 16;         // N offset within strip

    v8f acc[4] = {v8f{}, v8f{}, v8f{}, v8f{}};

    for (int k0 = 0; k0 < K; k0 += 32) {
        // ---- stage W[k0+sk, strip+sn0 .. +16) -> transposed LDS ----
        union { u32x4 q[2]; unsigned short s[16]; } wv;
        int kg = k0 + sk;
        if (kg < K) {
            const unsigned short* wp = W + (size_t)kg * N + tileN * 64 + sn0;
            wv.q[0] = *(const u32x4*)wp;
            wv.q[1] = *(const u32x4*)(wp + 8);
        } else {
#pragma unroll
            for (int j = 0; j < 16; ++j) wv.s[j] = 0;
        }
        __syncthreads();                      // previous iteration reads done
#pragma unroll
        for (int j = 0; j < 16; ++j)
            sBT[(sn0 + j) * 40 + sk] = wv.s[j];
        __syncthreads();

        __builtin_prefetch(A + (size_t)arow * lda + k0 + 32, 0, 1);

        // ---- A fragment: lanes 0-15 K{0..7,16..23}, lanes 16-31 K{8..15,24..31}
        union { v16bf v; u32x4 q[2]; } a;
        const unsigned short* ap = A + (size_t)arow * lda + k0 + half * 8;
        a.q[0] = *(const u32x4*)ap;
        a.q[1] = *(const u32x4*)(ap + 16);

        // ---- preload all 4 B fragments (distinct regs), single DS wait ----
        u32x4 bq[8];
#pragma unroll
        for (int s = 0; s < 4; ++s) {
            const unsigned short* bp = sBT + (s * 16 + l15) * 40 + half * 8;
            bq[2 * s]     = *(const u32x4*)bp;
            bq[2 * s + 1] = *(const u32x4*)(bp + 16);
        }
        // ---- 4 back-to-back WMMAs reusing the A fragment ----
#pragma unroll
        for (int s = 0; s < 4; ++s) {
            union { u32x4 q[2]; v16bf v; } b;
            b.q[0] = bq[2 * s];
            b.q[1] = bq[2 * s + 1];
            acc[s] = __builtin_amdgcn_wmma_f32_16x16x32_bf16(
                false, a.v, false, b.v, (short)0, acc[s], false, false);
        }
    }

#pragma unroll
    for (int s = 0; s < 4; ++s) {
#pragma unroll
        for (int r = 0; r < 8; ++r) {
            int row = tileM * 16 + half * 8 + r;
            int col = tileN * 64 + s * 16 + l15;
            float v = acc[s][r] + (bias ? bias[col] : 0.f);
            if (act == 1) v = tanhf(v);
            C[(size_t)row * N + col] = v;
        }
    }
}

// ---------------- small utility kernels ----------------
__global__ void gnn_fill_f32(float* p, float v, int n) {
    int i = blockIdx.x * blockDim.x + threadIdx.x;
    if (i < n) p[i] = v;
}
__global__ void gnn_init_edges(const int* si, const int* di, int* s, int* d, int* m) {
    int e = blockIdx.x * blockDim.x + threadIdx.x;
    if (e < ETOT) { s[e] = si[e]; d[e] = di[e]; m[e] = 1; }
}

// ---------------- attention (segment softmax over in-edges) ----------------
__global__ void gnn_att_alpha(const float* __restrict__ q, const float* __restrict__ k,
                              const float* __restrict__ eattr, const float* __restrict__ we,
                              const int* __restrict__ src, const int* __restrict__ dst,
                              const int* __restrict__ mask,
                              float* __restrict__ alpha, float* __restrict__ mrow, int n) {
    int t = blockIdx.x * blockDim.x + threadIdx.x;
    if (t >= ETOT * HH) return;
    int e = t >> 2, h = t & 3;
    if (!mask[e]) { alpha[t] = -1e30f; return; }
    int gid = e / EPG;
    int sg = gid * n + src[e];
    int dg = gid * n + dst[e];
    const float* qp = q + (size_t)dg * HC + h * CC;
    const float* kp = k + (size_t)sg * HC + h * CC;
    const float* er = eattr + (size_t)e * ED;
    float s = 0.f;
    for (int c = 0; c < CC; ++c)
        s += qp[c] * (kp[c] + edge_proj(er, we, h * CC + c));
    s *= 0.08838834764831843f;          // 1/sqrt(C)
    alpha[t] = s;
    atomicMaxF(&mrow[dg * HH + h], s);
}

__global__ void gnn_att_den(const float* __restrict__ alpha, const float* __restrict__ mrow,
                            const int* __restrict__ dst, const int* __restrict__ mask,
                            float* __restrict__ pe, float* __restrict__ den, int n) {
    int t = blockIdx.x * blockDim.x + threadIdx.x;
    if (t >= ETOT * HH) return;
    int e = t >> 2, h = t & 3;
    if (!mask[e]) { pe[t] = 0.f; return; }
    int gid = e / EPG;
    int dg = gid * n + dst[e];
    float p = __expf(alpha[t] - mrow[dg * HH + h]);
    pe[t] = p;
    atomicAdd(&den[dg * HH + h], p);
}

__global__ void gnn_att_agg(const float* __restrict__ pe, const float* __restrict__ den,
                            const float* __restrict__ v, const float* __restrict__ eattr,
                            const float* __restrict__ we,
                            const int* __restrict__ src, const int* __restrict__ dst,
                            const int* __restrict__ mask, float* __restrict__ out, int n) {
    int t = blockIdx.x * blockDim.x + threadIdx.x;
    if (t >= ETOT * HH) return;
    int e = t >> 2, h = t & 3;
    if (!mask[e]) return;
    int gid = e / EPG;
    int sg = gid * n + src[e];
    int dg = gid * n + dst[e];
    float dn = den[dg * HH + h];
    float a = pe[t] / (dn > 0.f ? dn : 1.f);
    if (a == 0.f) return;
    const float* vp = v + (size_t)sg * HC + h * CC;
    const float* er = eattr + (size_t)e * ED;
    for (int c = 0; c < CC; ++c)
        atomicAdd(&out[(size_t)dg * HC + h * CC + c],
                  (vp[c] + edge_proj(er, we, h * CC + c)) * a);
}

// ---------------- batch norm (training-mode batch stats) ----------------
__global__ void gnn_bn_stats(const float* __restrict__ x, float* mean, float* var, int M) {
    __shared__ float sh[256], sh2[256];
    int col = blockIdx.x;
    float s = 0.f, s2 = 0.f;
    for (int r = threadIdx.x; r < M; r += blockDim.x) {
        float v = x[(size_t)r * EMB + col];
        s += v; s2 += v * v;
    }
    sh[threadIdx.x] = s; sh2[threadIdx.x] = s2;
    __syncthreads();
    for (int off = 128; off > 0; off >>= 1) {
        if (threadIdx.x < off) { sh[threadIdx.x] += sh[threadIdx.x + off];
                                 sh2[threadIdx.x] += sh2[threadIdx.x + off]; }
        __syncthreads();
    }
    if (threadIdx.x == 0) {
        float mu = sh[0] / M;
        mean[col] = mu;
        var[col]  = sh2[0] / M - mu * mu;
    }
}
__global__ void gnn_bn_apply(float* x, const float* mean, const float* var,
                             const float* g, const float* b, int M) {
    int i = blockIdx.x * blockDim.x + threadIdx.x;
    if (i >= M * EMB) return;
    int col = i & (EMB - 1);
    x[i] = g[col] * (x[i] - mean[col]) * rsqrtf(var[col] + 1e-5f) + b[col];
}

// ---------------- TopK pooling (one block per graph) ----------------
__global__ void gnn_topk_pool(const float* __restrict__ x, const float* __restrict__ w,
                              float* __restrict__ xo, int* __restrict__ inv, int n, int k) {
    __shared__ float s[128], selv[128], red[128];
    __shared__ int taken[128], sel[128];
    __shared__ float wn;
    int g = blockIdx.x, t = threadIdx.x;
    float wv = w[t];
    red[t] = wv * wv;
    __syncthreads();
    for (int off = 64; off > 0; off >>= 1) {
        if (t < off) red[t] += red[t + off];
        __syncthreads();
    }
    if (t == 0) wn = sqrtf(red[0]);
    __syncthreads();
    if (t < n) {
        float d = 0.f;
        for (int c = 0; c < EMB; ++c) d += x[(size_t)(g * n + t) * EMB + c] * w[c];
        s[t] = tanhf(d / wn);
        taken[t] = 0;
    }
    __syncthreads();
    if (t == 0) {
        for (int j = 0; j < k; ++j) {
            float best = -2.f; int bi = 0;
            for (int i = 0; i < n; ++i)
                if (!taken[i] && s[i] > best) { best = s[i]; bi = i; }
            taken[bi] = 1; sel[j] = bi; selv[j] = best;
        }
    }
    __syncthreads();
    inv[g * 128 + t] = -1;
    __syncthreads();
    if (t < k) inv[g * 128 + sel[t]] = t;
    for (int j = 0; j < k; ++j)
        xo[(size_t)(g * k + j) * EMB + t] = x[(size_t)(g * n + sel[j]) * EMB + t] * selv[j];
}

__global__ void gnn_remap_edges(int* src, int* dst, int* mask, const int* inv) {
    int e = blockIdx.x * blockDim.x + threadIdx.x;
    if (e >= ETOT) return;
    int g = e / EPG;
    int ns = inv[g * 128 + src[e]];
    int nd = inv[g * 128 + dst[e]];
    mask[e] = mask[e] & (ns >= 0) & (nd >= 0);
    src[e] = ns > 0 ? ns : 0;
    dst[e] = nd > 0 ? nd : 0;
}

// ---------------- readout: h[g] += gmp || gap ----------------
__global__ void gnn_readout(const float* __restrict__ x, float* __restrict__ h, int n) {
    int g = blockIdx.x, c = threadIdx.x;
    float mx = -1e30f, sm = 0.f;
    for (int r = 0; r < n; ++r) {
        float v = x[(size_t)(g * n + r) * EMB + c];
        mx = fmaxf(mx, v); sm += v;
    }
    h[g * 256 + c]       += mx;
    h[g * 256 + 128 + c] += sm / n;
}

__global__ void gnn_final_l3(const float* h2, const float* w, const float* b, float* out) {
    int g = threadIdx.x;
    float s = 0.f;
    for (int i = 0; i < 64; ++i) s += h2[g * 64 + i] * w[i];
    out[g] = s + b[0];
}

// ---------------- host orchestration ----------------
extern "C" void kernel_launch(void* const* d_in, const int* in_sizes, int n_in,
                              void* d_out, int out_size, void* d_ws, size_t ws_size,
                              hipStream_t stream) {
    auto P = [&](int i) { return (const float*)d_in[i]; };
    const float* d_x     = P(0);
    const float* d_eattr = P(1);
    const int* d_src = (const int*)d_in[35];
    const int* d_dst = (const int*)d_in[36];

    char* wp = (char*)d_ws;
    auto carve = [&](size_t bytes) {
        char* r = wp; wp += (bytes + 255) & ~(size_t)255; return (void*)r;
    };
    float* f_q   = (float*)carve(sizeof(float) * NTOT * HC);
    float* f_k   = (float*)carve(sizeof(float) * NTOT * HC);
    float* f_v   = (float*)carve(sizeof(float) * NTOT * HC);
    float* f_o   = (float*)carve(sizeof(float) * NTOT * HC);
    float* f_al  = (float*)carve(sizeof(float) * ETOT * HH);
    float* f_pe  = (float*)carve(sizeof(float) * ETOT * HH);
    float* f_m   = (float*)carve(sizeof(float) * NTOT * HH);
    float* f_den = (float*)carve(sizeof(float) * NTOT * HH);
    float* xA    = (float*)carve(sizeof(float) * NTOT * EMB);
    float* xB    = (float*)carve(sizeof(float) * NTOT * EMB);
    float* f_mu  = (float*)carve(sizeof(float) * EMB);
    float* f_va  = (float*)carve(sizeof(float) * EMB);
    float* f_h   = (float*)carve(sizeof(float) * BGR * 256);
    float* f_h1  = (float*)carve(sizeof(float) * BGR * 256);
    float* f_h2  = (float*)carve(sizeof(float) * BGR * 64);
    int* i_src   = (int*)carve(sizeof(int) * ETOT);
    int* i_dst   = (int*)carve(sizeof(int) * ETOT);
    int* i_msk   = (int*)carve(sizeof(int) * ETOT);
    int* i_inv   = (int*)carve(sizeof(int) * BGR * 128);
    unsigned short* a_act = (unsigned short*)carve(sizeof(short) * NTOT * HC);
    unsigned short* wb_c1q = (unsigned short*)carve(sizeof(short) * F0 * HC);
    unsigned short* wb_c1k = (unsigned short*)carve(sizeof(short) * F0 * HC);
    unsigned short* wb_c1v = (unsigned short*)carve(sizeof(short) * F0 * HC);
    unsigned short* wb_c1s = (unsigned short*)carve(sizeof(short) * F0 * HC);
    unsigned short* wb_cq  = (unsigned short*)carve(sizeof(short) * 9 * EMB * HC);
    unsigned short* wb_ck  = (unsigned short*)carve(sizeof(short) * 9 * EMB * HC);
    unsigned short* wb_cv  = (unsigned short*)carve(sizeof(short) * 9 * EMB * HC);
    unsigned short* wb_cs  = (unsigned short*)carve(sizeof(short) * 9 * EMB * HC);
    unsigned short* wb_t1  = (unsigned short*)carve(sizeof(short) * HC * EMB);
    unsigned short* wb_t   = (unsigned short*)carve(sizeof(short) * 9 * HC * EMB);
    unsigned short* wb_l1  = (unsigned short*)carve(sizeof(short) * 256 * 256);
    unsigned short* wb_l2  = (unsigned short*)carve(sizeof(short) * 256 * 64);

    auto cvt = [&](const float* s, unsigned short* d, int n) {
        gnn_cvt_bf16<<<(n + 255) / 256, 256, 0, stream>>>(s, d, n);
    };
    auto cvt_pad = [&](const float* s, unsigned short* d, int M, int K, int Kp) {
        gnn_cvt_pad_bf16<<<(M * Kp + 255) / 256, 256, 0, stream>>>(s, d, M, K, Kp);
    };
    auto gemm = [&](const unsigned short* A, int lda, const unsigned short* W,
                    const float* bias, float* C, int M, int K, int N, int act) {
        dim3 blk(32, 4), grd(N / 64, M / 64);
        gnn_wmma_gemm_bf16<<<grd, blk, 0, stream>>>(A, lda, W, bias, C, M, K, N, act);
    };
    auto fill = [&](float* p, float v, int n) {
        gnn_fill_f32<<<(n + 255) / 256, 256, 0, stream>>>(p, v, n);
    };

    // one-time weight conversions (outside all hot loops)
    cvt(P(2),  wb_c1q, F0 * HC);  cvt(P(4),  wb_c1k, F0 * HC);
    cvt(P(6),  wb_c1v, F0 * HC);  cvt(P(9),  wb_c1s, F0 * HC);
    cvt(P(11), wb_cq, 9 * EMB * HC); cvt(P(13), wb_ck, 9 * EMB * HC);
    cvt(P(15), wb_cv, 9 * EMB * HC); cvt(P(18), wb_cs, 9 * EMB * HC);
    cvt(P(20), wb_t1, HC * EMB);     cvt(P(22), wb_t, 9 * HC * EMB);
    cvt(P(29), wb_l1, 256 * 256);    cvt(P(31), wb_l2, 256 * 64);

    gnn_init_edges<<<(ETOT + 255) / 256, 256, 0, stream>>>(d_src, d_dst, i_src, i_dst, i_msk);
    fill(f_h, 0.f, BGR * 256);

    int n = NN0;
    float* xcur = nullptr;
    const int EHgrid = (ETOT * HH + 255) / 256;

    for (int L = 0; L < 10; ++L) {
        const unsigned short *wq, *wk, *wv, *wsk, *tw;
        const float *bq, *bk, *bv, *we, *bs, *tb, *bg, *bb;
        int Kin, Kp;
        if (L == 0) {
            wq = wb_c1q; wk = wb_c1k; wv = wb_c1v; wsk = wb_c1s;
            bq = P(3); bk = P(5); bv = P(7); we = P(8); bs = P(10);
            tw = wb_t1; tb = P(21); bg = P(24); bb = P(25);
            Kin = F0; Kp = 32;
        } else {
            int i = L - 1;
            wq = wb_cq + (size_t)i * EMB * HC; wk = wb_ck + (size_t)i * EMB * HC;
            wv = wb_cv + (size_t)i * EMB * HC; wsk = wb_cs + (size_t)i * EMB * HC;
            bq = P(12) + (size_t)i * HC; bk = P(14) + (size_t)i * HC;
            bv = P(16) + (size_t)i * HC; bs = P(19) + (size_t)i * HC;
            we = P(17) + (size_t)i * ED * HC;
            tw = wb_t + (size_t)i * HC * EMB; tb = P(23) + (size_t)i * EMB;
            bg = P(26) + (size_t)i * EMB; bb = P(27) + (size_t)i * EMB;
            Kin = EMB; Kp = EMB;
        }
        const float* Af = (L == 0) ? d_x : xcur;
        int M = BGR * n;

        cvt_pad(Af, a_act, M, Kin, Kp);
        gemm(a_act, Kp, wq,  bq, f_q, M, Kin, HC, 0);
        gemm(a_act, Kp, wk,  bk, f_k, M, Kin, HC, 0);
        gemm(a_act, Kp, wv,  bv, f_v, M, Kin, HC, 0);
        gemm(a_act, Kp, wsk, bs, f_o, M, Kin, HC, 0);   // out = skip

        fill(f_m, -1e30f, M * HH);
        fill(f_den, 0.f, M * HH);
        gnn_att_alpha<<<EHgrid, 256, 0, stream>>>(f_q, f_k, d_eattr, we,
                                                  i_src, i_dst, i_msk, f_al, f_m, n);
        gnn_att_den<<<EHgrid, 256, 0, stream>>>(f_al, f_m, i_dst, i_msk, f_pe, f_den, n);
        gnn_att_agg<<<EHgrid, 256, 0, stream>>>(f_pe, f_den, f_v, d_eattr, we,
                                                i_src, i_dst, i_msk, f_o, n);

        float* xnext = (xcur == xA) ? xB : xA;
        cvt_pad(f_o, a_act, M, HC, HC);
        gemm(a_act, HC, tw, tb, xnext, M, HC, EMB, 0);
        gnn_bn_stats<<<EMB, 256, 0, stream>>>(xnext, f_mu, f_va, M);
        gnn_bn_apply<<<(M * EMB + 255) / 256, 256, 0, stream>>>(xnext, f_mu, f_va, bg, bb, M);
        xcur = xnext;

        if (L >= 1 && (L - 1) % 3 == 0) {
            int i = L - 1;
            int k = (i == 0) ? 109 : (i == 3) ? 93 : 80;
            const float* pw = P(28) + (size_t)(i / 3) * EMB;
            float* xp = (xcur == xA) ? xB : xA;
            gnn_topk_pool<<<BGR, 128, 0, stream>>>(xcur, pw, xp, i_inv, n, k);
            gnn_remap_edges<<<(ETOT + 255) / 256, 256, 0, stream>>>(i_src, i_dst, i_msk, i_inv);
            n = k; xcur = xp;
            gnn_readout<<<BGR, 128, 0, stream>>>(xcur, f_h, n);
        }
    }

    // MLP head
    cvt_pad(f_h, a_act, BGR, 256, 256);
    gemm(a_act, 256, wb_l1, P(30), f_h1, BGR, 256, 256, 1);
    cvt_pad(f_h1, a_act, BGR, 256, 256);
    gemm(a_act, 256, wb_l2, P(32), f_h2, BGR, 256, 64, 1);
    gnn_final_l3<<<1, 64, 0, stream>>>(f_h2, P(33), P(34), (float*)d_out);
    (void)in_sizes; (void)n_in; (void)out_size; (void)ws_size;
}
